// CharCNNWordEncoder_90177133347585
// MI455X (gfx1250) — matrix operand order repaired and checked by reading, not compiled, not run on hardware
//
#include <hip/hip_runtime.h>
#include <hip/hip_bf16.h>

// ---------------------------------------------------------------------------
// CharCNN word encoder for gfx1250 (MI455X), fused as ONE f16 WMMA GEMM:
//   rows  = (word, char_pos) pairs : 65536*20
//   K     = 5 taps x 8 channels = 40, +1 mask-indicator slot (K=40), pad to 64
//   cols  = 150 conv outputs (k=3,4,5 x 50), padded to 160
// The char-position mask is folded INTO the GEMM: A[row][40] = !valid(row),
// Wc[40][*] = -30000, so invalid positions emerge 30000 below any valid conv
// value and the max-pool needs no per-element selects.
// Matrix op: v_wmma_f32_16x16x32_f16 (f32 accumulate).
// ---------------------------------------------------------------------------

typedef __attribute__((ext_vector_type(16))) _Float16 v16h;
typedef __attribute__((ext_vector_type(8)))  _Float16 v8h;
typedef __attribute__((ext_vector_type(8)))  float    v8f;

union Frag16 { v16h h16; float4 f4[2]; };
union Half4  { v8h h8; float4 f4; };

#define NB        512
#define NW        128
#define NC        20
#define NWORDS    (NB * NW)       // 65536
#define INCH      8
#define OUTCH     50
#define NCOLS     150
#define NCOLS_PAD 160
#define KPAD      64
#define BWORDS    64              // words per workgroup
#define WPW       8               // words per wave (8 waves * 8 = 64)
#define MASKPEN   (-30000.0f)     // exactly representable in f16 (step 16)
#define LOWF      (-1.0e30f)

// ------------------------- 1) mask prefix scan -----------------------------
// Single workgroup exclusive scan over the 65536 word-mask bytes -> compact
// index for every valid word (stable valid-first order == argsort(!mask)).
__global__ __launch_bounds__(1024) void scan_mask_kernel(
    const unsigned char* __restrict__ wmask, int* __restrict__ cidx, int n) {
  __shared__ int sums[1024];
  const int t = threadIdx.x;
  const int per = n / 1024;               // 64
  const int base = t * per;
  int cnt = 0;
  for (int i = 0; i < per; ++i) cnt += wmask[base + i] ? 1 : 0;
  sums[t] = cnt;
  __syncthreads();
  for (int off = 1; off < 1024; off <<= 1) {
    int v = 0;
    if (t >= off) v = sums[t - off];
    __syncthreads();
    sums[t] += v;
    __syncthreads();
  }
  int run = sums[t] - cnt;                // exclusive prefix
  for (int i = 0; i < per; ++i) {
    cidx[base + i] = run;
    run += wmask[base + i] ? 1 : 0;
  }
}

// ------------------------- 2) fused WMMA conv+maxpool ----------------------
__global__ __launch_bounds__(256) void charcnn_wmma_kernel(
    const int* __restrict__ chars,            // [NWORDS][20]
    const unsigned char* __restrict__ wmask,  // [NWORDS]
    const unsigned char* __restrict__ cmask,  // [NWORDS][20]
    const float* __restrict__ emb,            // [262][8]
    const float* __restrict__ w3, const float* __restrict__ b3,
    const float* __restrict__ w4, const float* __restrict__ b4,
    const float* __restrict__ w5, const float* __restrict__ b5,
    const int* __restrict__ cidx,             // [NWORDS]
    float* __restrict__ featsC)               // [n_true][150] (compacted)
{
  // Zero-padded f16 embeddings: 24 positions (2 pad each side) x 8 ch.
  __shared__ alignas(16) _Float16 s_emb[BWORDS][24][INCH];      // 24576 B
  // Combined weights, K-contiguous per column: s_w[col][kk].
  __shared__ alignas(16) _Float16 s_w[NCOLS_PAD][KPAD];         // 20480 B
  __shared__ unsigned int s_mbits[BWORDS];                      //   256 B

  const int tid = threadIdx.x;
  const int wgBase = blockIdx.x * BWORDS;

  // Stage embeddings (f32 -> f16) with zero padding.
  for (int idx = tid; idx < BWORDS * 24; idx += 256) {
    const int w = idx / 24, p = idx % 24;
    v8h row;
#pragma unroll
    for (int i = 0; i < INCH; ++i) row[i] = (_Float16)0.0f;
    if (p >= 2 && p < 22) {
      const int c = chars[(wgBase + w) * NC + (p - 2)];
      const float* e = emb + c * INCH;
#pragma unroll
      for (int i = 0; i < INCH; ++i) row[i] = (_Float16)e[i];
    }
    *(v8h*)&s_emb[w][p][0] = row;
  }

  // Build combined weight matrix Wc (transposed: [col][K]).
  // K = tap*8 + ch, tap in 0..4 (window t-2..t+2).
  //   cols [0,50)   : conv3 at taps 1..3  (pad_left=1)
  //   cols [50,100) : conv4 at taps 0..3  (pad_left=2)
  //   cols [100,150): conv5 at taps 0..4  (pad_left=2)
  // K==40 : mask-penalty row (A carries the !valid indicator there).
  for (int idx = tid; idx < NCOLS_PAD * KPAD; idx += 256) {
    const int col = idx >> 6, kk = idx & 63;
    const int tap = kk >> 3, ch = kk & 7;
    float v = 0.0f;
    if (kk < 40) {
      if (col < 50) {
        const int dk = tap - 1;
        if (dk >= 0 && dk < 3) v = w3[(col * INCH + ch) * 3 + dk];
      } else if (col < 100) {
        if (tap < 4) v = w4[((col - 50) * INCH + ch) * 4 + tap];
      } else if (col < 150) {
        v = w5[((col - 100) * INCH + ch) * 5 + tap];
      }
    } else if (kk == 40) {
      v = MASKPEN;
    }
    s_w[col][kk] = (_Float16)v;
  }

  // Per-word char-validity bitmaps (20 bits).
  for (int w = tid; w < BWORDS; w += 256) {
    unsigned int bits = 0;
    const unsigned char* m = cmask + (size_t)(wgBase + w) * NC;
    for (int p = 0; p < NC; ++p) bits |= (m[p] ? 1u : 0u) << p;
    s_mbits[w] = bits;
  }
  __syncthreads();

  const int wave = tid >> 5;
  const int lane = tid & 31;
  const int g    = lane >> 4;     // lane half (selects K sub-range)
  const int ln   = lane & 15;     // M row (A) / N col (B,C,D)
  const int waveWord = wave * WPW;

  // Destination compact index per word (-1 if masked-out word).
  int dstw[WPW];
#pragma unroll
  for (int w = 0; w < WPW; ++w) {
    const int n = wgBase + waveWord + w;
    dstw[w] = wmask[n] ? cidx[n] : -1;
  }

  const float4 z4 = make_float4(0.f, 0.f, 0.f, 0.f);

  // ------- Precompute ALL per-m-tile A data before the nt loop (so the hot
  // ------- loop contains only B loads, WMMAs and the max epilogue). -------
  Frag16 AF0[10];     // taps {0,2}/{1,3} per lane half  (K 0..31)
  float4 a1lo[10];    // low half of AF1: g=0 -> tap4 (K 32..39);
                      //                  g=1 -> {indicator@K40, 0...}
#pragma unroll
  for (int mt = 0; mt < 10; ++mt) {
    const int RB    = mt * 16;
    const int wA    = RB / 20;
    const int bndry = (wA + 1) * 20;
    const int row   = RB + ln;                 // A row for ALL lanes
    const int wloc  = (row >= bndry) ? (wA + 1) : wA;
    const int pos   = row - wloc * 20;
    const _Float16* erow = &s_emb[waveWord + wloc][pos][0];
    const int t0 = g ? 1 : 0;                  // g=0 taps {0,2}; g=1 taps {1,3}
    AF0[mt].f4[0] = *(const float4*)(erow + t0 * 8);
    AF0[mt].f4[1] = *(const float4*)(erow + (t0 + 2) * 8);
    const float4 tap4 = *(const float4*)(erow + 4 * 8);
    const unsigned int vb = (s_mbits[waveWord + wloc] >> pos) & 1u;
    Half4 indv;
    indv.f4 = z4;
    indv.h8[0] = vb ? (_Float16)0.0f : (_Float16)1.0f;  // K=40 indicator
    a1lo[mt] = g ? indv.f4 : tap4;             // branchless select
  }

  for (int nt = 0; nt < 10; ++nt) {
    // B fragments: element j<8 = Wc[s*32 + g*8 + j], j>=8 = Wc[s*32 + 16 + g*8 + j-8] (col nt*16 + ln).
    Frag16 bf0, bf1;
    {
      const _Float16* wrow = &s_w[nt * 16 + ln][0];
      const int base0 = g * 8;
      bf0.f4[0] = *(const float4*)(wrow + base0);
      bf0.f4[1] = *(const float4*)(wrow + base0 + 16);
      bf1.f4[0] = *(const float4*)(wrow + 32 + base0);
      bf1.f4[1] = *(const float4*)(wrow + 32 + base0 + 16);
    }

    float wmax[WPW];
#pragma unroll
    for (int w = 0; w < WPW; ++w) wmax[w] = LOWF;

#pragma unroll
    for (int mt = 0; mt < 10; ++mt) {
      const int RB    = mt * 16;        // compile-time after unroll
      const int wA    = RB / 20;        // first word touched by this tile
      const int bndry = (wA + 1) * 20;  // first row of word wA+1
      const int s     = bndry - RB;     // split point in tile (4/8/12/>=16)

      Frag16 af1;
      af1.f4[0] = a1lo[mt];
      af1.f4[1] = z4;

      // --- 16x16 tile: D = A0*B0 + A1*B1 (mask penalty included) ---
      v8f c = {};
      c = __builtin_amdgcn_wmma_f32_16x16x32_f16(
              false, AF0[mt].h16, false, bf0.h16, (short)0, c, false, false);
      c = __builtin_amdgcn_wmma_f32_16x16x32_f16(
              false, af1.h16, false, bf1.h16, (short)0, c, false, false);

      // --- max over tile rows (D: M = r + g*8, N = ln), split-aware ---
      if (s >= 16) {
        // tile entirely inside word wA
        float pm = fmaxf(fmaxf(fmaxf(c[0], c[1]), fmaxf(c[2], c[3])),
                         fmaxf(fmaxf(c[4], c[5]), fmaxf(c[6], c[7])));
        pm = fmaxf(pm, __shfl_xor(pm, 16, 32));
        wmax[wA] = fmaxf(wmax[wA], pm);
      } else if (s == 8) {
        // split exactly at the lane-half boundary: g=0 all wA, g=1 all wA+1
        float pm = fmaxf(fmaxf(fmaxf(c[0], c[1]), fmaxf(c[2], c[3])),
                         fmaxf(fmaxf(c[4], c[5]), fmaxf(c[6], c[7])));
        const float other = __shfl_xor(pm, 16, 32);
        const float pA = g ? other : pm;
        const float pB = g ? pm : other;
        wmax[wA] = fmaxf(wmax[wA], pA);
        if (wA + 1 < WPW) wmax[wA + 1] = fmaxf(wmax[wA + 1], pB);
      } else {
        // s==4 : g=0 splits 4|4, g=1 entirely wA+1
        // s==12: g=0 entirely wA, g=1 splits 4|4
        const float mlo = fmaxf(fmaxf(c[0], c[1]), fmaxf(c[2], c[3]));
        const float mhi = fmaxf(fmaxf(c[4], c[5]), fmaxf(c[6], c[7]));
        const float mall = fmaxf(mlo, mhi);
        float ownA, ownB;
        if (s == 4) { ownA = g ? LOWF : mlo;  ownB = g ? mall : mhi; }
        else        { ownA = g ? mlo  : mall; ownB = g ? mhi  : LOWF; }
        const float oA = __shfl_xor(ownA, 16, 32);
        const float oB = __shfl_xor(ownB, 16, 32);
        wmax[wA] = fmaxf(wmax[wA], fmaxf(ownA, oA));
        if (wA + 1 < WPW) wmax[wA + 1] = fmaxf(wmax[wA + 1], fmaxf(ownB, oB));
      }
    }

    // --- store: bias + compacted scatter (lanes g==0 own distinct cols) ---
    const int colg = nt * 16 + ln;
    if (g == 0 && colg < NCOLS) {
      const float bias = (colg < 50) ? b3[colg]
                       : (colg < 100) ? b4[colg - 50]
                                      : b5[colg - 100];
#pragma unroll
      for (int w = 0; w < WPW; ++w) {
        if (dstw[w] >= 0) {
          featsC[(size_t)dstw[w] * NCOLS + colg] = wmax[w] + bias;
        }
      }
    }
  }
}

// ------------------------- 3) words_id gather ------------------------------
__global__ __launch_bounds__(160) void gather_kernel(
    const int* __restrict__ words_id, const float* __restrict__ featsC,
    float* __restrict__ out) {
  const int n = blockIdx.x;        // word slot 0..65535
  const int f = threadIdx.x;
  if (f < NCOLS) {
    const int id = words_id[n];    // compact index (reference: ids < n_true)
    out[(size_t)n * NCOLS + f] = featsC[(size_t)id * NCOLS + f];
  }
}

// ---------------------------------------------------------------------------
extern "C" void kernel_launch(void* const* d_in, const int* in_sizes, int n_in,
                              void* d_out, int out_size, void* d_ws, size_t ws_size,
                              hipStream_t stream) {
  (void)in_sizes; (void)n_in; (void)out_size; (void)ws_size;
  const int*           chars = (const int*)d_in[0];
  const unsigned char* wmask = (const unsigned char*)d_in[1];
  const unsigned char* cmask = (const unsigned char*)d_in[2];
  const int*           wid   = (const int*)d_in[3];
  const float*         emb   = (const float*)d_in[4];
  const float* w3 = (const float*)d_in[5];
  const float* b3 = (const float*)d_in[6];
  const float* w4 = (const float*)d_in[7];
  const float* b4 = (const float*)d_in[8];
  const float* w5 = (const float*)d_in[9];
  const float* b5 = (const float*)d_in[10];
  float* out = (float*)d_out;

  // Workspace layout: [cidx: 65536 ints][featsC: n_true*150 f32]
  int*   cidx   = (int*)d_ws;
  size_t off    = ((size_t)NWORDS * sizeof(int) + 255) & ~(size_t)255;
  float* featsC = (float*)((char*)d_ws + off);

  scan_mask_kernel<<<1, 1024, 0, stream>>>(wmask, cidx, NWORDS);
  charcnn_wmma_kernel<<<NWORDS / BWORDS, 256, 0, stream>>>(
      chars, wmask, cmask, emb, w3, b3, w4, b4, w5, b5, cidx, featsC);
  gather_kernel<<<NWORDS, 160, 0, stream>>>(wid, featsC, out);
}
